// RetroFPN_52218212384897
// MI455X (gfx1250) — matrix operand order repaired and checked by reading, not compiled, not run on hardware
//
#include <hip/hip_runtime.h>
#include <math.h>

// ---------------- types ----------------
typedef __attribute__((ext_vector_type(16))) __bf16        v16bf;
typedef __attribute__((ext_vector_type(16))) unsigned short v16u;
typedef __attribute__((ext_vector_type(8)))  float          v8f;

typedef unsigned short bf16_t;   // storage type (host-safe)

#define NPTS 100000
#define CCH  96
#define GRP  6
#define NBR  16
#define BNEPS 1e-5f

// fp32 <-> bf16 via native hardware conversion (v_cvt_*bf16_f32 on gfx1250)
__device__ __forceinline__ bf16_t f2bf(float f) {
    __bf16 h = (__bf16)f;
    return __builtin_bit_cast(bf16_t, h);
}
__device__ __forceinline__ float bf2f(bf16_t s) {
    return (float)__builtin_bit_cast(__bf16, s);
}
__device__ __forceinline__ v16bf as_bf(v16u u) { return __builtin_bit_cast(v16bf, u); }

// ---------------------------------------------------------------------------
// Kernel 0: pack weights to bf16 in WMMA B-fragment order.
// B fragment (16x16x32 bf16): lane holds column n = lane&15, K = (lane>>4)*8 + e + (e<8?0:8),
// packed as pack[(kt*NT+nt)*512 + lane*16 + e]  (e contiguous in K -> coalesced)
// ---------------------------------------------------------------------------
__global__ void pack_weights(const float* qw, const float* kw, const float* vw,
                             const float* p2w, const float* w1w,
                             bf16_t* qkvp, bf16_t* p2p, bf16_t* w1p) {
    int tid   = blockIdx.x * blockDim.x + threadIdx.x;
    int tot   = blockDim.x * gridDim.x;
    // 4 big 96x96 matrices: q_w, k_w, v_w, p2_w : each 3 kt * 6 nt * 512 = 9216
    for (int p = tid; p < 4 * 9216; p += tot) {
        int mat = p / 9216;
        int q   = p % 9216;
        int kt  = q / (6 * 512);
        int rem = q % (6 * 512);
        int nt  = rem / 512;
        int lq  = rem % 512;
        int lane = lq / 16, e = lq % 16;
        int kg = kt * 32 + (lane >> 4) * 8 + e + (e < 8 ? 0 : 8);
        int ng = nt * 16 + (lane & 15);
        const float* src = (mat == 0) ? qw : (mat == 1) ? kw : (mat == 2) ? vw : p2w;
        bf16_t* dst = (mat < 3) ? (qkvp + mat * 9216) : p2p;
        dst[q] = f2bf(src[kg * CCH + ng]);
    }
    // w1_w [96,6] -> padded to [96,16]: 3 kt * 512 = 1536
    for (int p = tid; p < 3 * 512; p += tot) {
        int kt = p / 512;
        int lq = p % 512;
        int lane = lq / 16, e = lq % 16;
        int kg = kt * 32 + (lane >> 4) * 8 + e + (e < 8 ? 0 : 8);
        int ng = lane & 15;
        float v = (ng < GRP) ? w1w[kg * GRP + ng] : 0.0f;
        w1p[p] = f2bf(v);
    }
}

// ---------------------------------------------------------------------------
// Kernel 1: q/k/v = act(bn(feat @ W + b)).  One wave per 16-row tile.
// A fragment 16-bit 16x32: lane m=lane&15, half=lane>>4,
//   local K = e + (e<8?0:8) + 8*half  => two contiguous 8-float runs per kt.
// ---------------------------------------------------------------------------
__global__ __launch_bounds__(128)
void qkv_kernel(const float* __restrict__ feat,
                const float* qb, const float* qg, const float* qbeta,
                const float* kb, const float* kgam, const float* kbeta,
                const float* vb,
                const bf16_t* __restrict__ qkvp,
                bf16_t* __restrict__ qbuf, bf16_t* __restrict__ kbuf,
                bf16_t* __restrict__ vbuf) {
    int wave = threadIdx.x >> 5;
    int lane = threadIdx.x & 31;
    int tile = blockIdx.x * (blockDim.x >> 5) + wave;
    if (tile * 16 >= NPTS) return;
    int m = lane & 15, half = lane >> 4;
    int row = tile * 16 + m;

    // build A fragments from feat (fp32 -> bf16)
    v16bf afrag[3];
    const float* fr = feat + (size_t)row * CCH;
    #pragma unroll
    for (int kt = 0; kt < 3; ++kt) {
        int c0 = kt * 32 + 8 * half;
        v16u au;
        #pragma unroll
        for (int e = 0; e < 8; ++e) au[e]     = f2bf(fr[c0 + e]);
        #pragma unroll
        for (int e = 0; e < 8; ++e) au[8 + e] = f2bf(fr[c0 + 16 + e]);
        afrag[kt] = as_bf(au);
    }

    #pragma unroll
    for (int mat = 0; mat < 3; ++mat) {
        const bf16_t* wp = qkvp + mat * 9216;
        bf16_t* ob = (mat == 0) ? qbuf : (mat == 1) ? kbuf : vbuf;
        for (int nt = 0; nt < 6; ++nt) {
            v8f acc = {};
            #pragma unroll
            for (int kt = 0; kt < 3; ++kt) {
                v16u bu = *(const v16u*)(wp + (kt * 6 + nt) * 512 + lane * 16);
                acc = __builtin_amdgcn_wmma_f32_16x16x32_bf16(
                        false, afrag[kt], false, as_bf(bu), (short)0, acc, false, false);
            }
            int c = nt * 16 + m;
            float bias, scale = 1.0f, shift = 0.0f;
            float bnr = rsqrtf(1.0f + BNEPS);
            if (mat == 0)      { bias = qb[c]; scale = qg[c] * bnr;  shift = qbeta[c]; }
            else if (mat == 1) { bias = kb[c]; scale = kgam[c] * bnr; shift = kbeta[c]; }
            else               { bias = vb[c]; }
            #pragma unroll
            for (int r = 0; r < 8; ++r) {
                int rr = tile * 16 + r + 8 * half;
                float vv = acc[r] + bias;
                if (mat < 2) { vv = vv * scale + shift; vv = vv > 0.f ? vv : 0.f; }
                ob[(size_t)rr * CCH + c] = f2bf(vv);
            }
        }
    }
}

// ---------------------------------------------------------------------------
// Kernel 2: full per-point attention. One wave per point (S=16 == one M tile).
// ---------------------------------------------------------------------------
__global__ __launch_bounds__(128)
void attn_kernel(const float* __restrict__ coord, const int* __restrict__ refidx,
                 const float* p1w, const float* p1b, const float* pg, const float* pbeta,
                 const float* p2b,
                 const float* w1b, const float* wgam, const float* wbeta,
                 const float* w2w, const float* w2b,
                 const bf16_t* __restrict__ p2p, const bf16_t* __restrict__ w1p,
                 const bf16_t* __restrict__ qbuf, const bf16_t* __restrict__ kbuf,
                 const bf16_t* __restrict__ vbuf,
                 float* __restrict__ out) {
    __shared__ float peb_lds[4][NBR * CCH];   // 24 KB
    __shared__ float larr[4][NBR * GRP];
    __shared__ float wgt_lds[4][NBR * GRP];
    __shared__ int   nbl[4][NBR];

    int wave = threadIdx.x >> 5;
    int lane = threadIdx.x & 31;
    int n = blockIdx.x * (blockDim.x >> 5) + wave;
    if (n >= NPTS) return;
    int m = lane & 15, half = lane >> 4;
    float bnr = rsqrtf(1.0f + BNEPS);

    // neighbour index + relative position for row m
    int nb = refidx[(size_t)n * NBR + m];
    if (half == 0) nbl[wave][m] = nb;
    int nbc = nb < 0 ? 0 : nb;
    float px = coord[(size_t)nbc * 3 + 0] - coord[(size_t)n * 3 + 0];
    float py = coord[(size_t)nbc * 3 + 1] - coord[(size_t)n * 3 + 1];
    float pz = coord[(size_t)nbc * 3 + 2] - coord[(size_t)n * 3 + 2];

    // hidden h = relu(bn(pos @ p1_w + p1_b)) built directly in A-fragment layout
    v16bf hfrag[3];
    #pragma unroll
    for (int kt = 0; kt < 3; ++kt) {
        int c0 = kt * 32 + 8 * half;
        v16u hu;
        #pragma unroll
        for (int e = 0; e < 16; ++e) {
            int c = c0 + e + (e < 8 ? 0 : 8);
            float hv = px * p1w[c] + py * p1w[CCH + c] + pz * p1w[2 * CCH + c] + p1b[c];
            hv = hv * (pg[c] * bnr) + pbeta[c];
            hv = hv > 0.f ? hv : 0.f;
            hu[e] = f2bf(hv);
        }
        hfrag[kt] = as_bf(hu);
    }

    // peb = h @ p2_w + p2_b  -> LDS (C-fragment -> row-major transpose)
    float* peb = peb_lds[wave];
    for (int nt = 0; nt < 6; ++nt) {
        v8f acc = {};
        #pragma unroll
        for (int kt = 0; kt < 3; ++kt) {
            v16u bu = *(const v16u*)(p2p + (kt * 6 + nt) * 512 + lane * 16);
            acc = __builtin_amdgcn_wmma_f32_16x16x32_bf16(
                    false, hfrag[kt], false, as_bf(bu), (short)0, acc, false, false);
        }
        int c = nt * 16 + m;
        float bias = p2b[c];
        #pragma unroll
        for (int r = 0; r < 8; ++r)
            peb[(r + 8 * half) * CCH + c] = acc[r] + bias;
    }
    asm volatile("s_wait_dscnt 0x0" ::: "memory");

    // rel = k[idx] - q + peb  (bf16 gathers hit L2; two contiguous 8-elem runs)
    v16bf rfrag[3];
    const bf16_t* qrow = qbuf + (size_t)n * CCH;
    const bf16_t* krow = kbuf + (size_t)nbc * CCH;
    #pragma unroll
    for (int kt = 0; kt < 3; ++kt) {
        int c0 = kt * 32 + 8 * half;
        v16u ru;
        #pragma unroll
        for (int e = 0; e < 16; ++e) {
            int c = c0 + e + (e < 8 ? 0 : 8);
            float rv = bf2f(krow[c]) - bf2f(qrow[c]) + peb[m * CCH + c];
            ru[e] = f2bf(rv);
        }
        rfrag[kt] = as_bf(ru);
    }

    // wgt logits = rel @ w1 (padded to 16 cols, 6 valid)
    v8f accw = {};
    #pragma unroll
    for (int kt = 0; kt < 3; ++kt) {
        v16u bu = *(const v16u*)(w1p + kt * 512 + lane * 16);
        accw = __builtin_amdgcn_wmma_f32_16x16x32_bf16(
                false, rfrag[kt], false, as_bf(bu), (short)0, accw, false, false);
    }
    float* la = larr[wave];
    if (m < GRP) {
        float scale = wgam[m] * bnr, shift = wbeta[m], bias = w1b[m];
        #pragma unroll
        for (int r = 0; r < 8; ++r) {
            float v = (accw[r] + bias) * scale + shift;
            la[(r + 8 * half) * GRP + m] = v > 0.f ? v : 0.f;
        }
    }
    asm volatile("s_wait_dscnt 0x0" ::: "memory");

    // second GxG linear + softmax over neighbours + padding mask (lanes 0..5)
    float* wl = wgt_lds[wave];
    if (lane < GRP) {
        int g = lane;
        float t[NBR], mx = -1e30f;
        for (int s = 0; s < NBR; ++s) {
            float a2 = w2b[g];
            #pragma unroll
            for (int gp = 0; gp < GRP; ++gp) a2 += la[s * GRP + gp] * w2w[gp * GRP + g];
            t[s] = a2;
            mx = a2 > mx ? a2 : mx;
        }
        float sum = 0.f;
        for (int s = 0; s < NBR; ++s) { t[s] = __expf(t[s] - mx); sum += t[s]; }
        float inv = 1.0f / sum;
        for (int s = 0; s < NBR; ++s) {
            float msk = (nbl[wave][s] >= 0) ? 1.0f : 0.0f;
            wl[s * GRP + g] = t[s] * inv * msk;
        }
    }
    asm volatile("s_wait_dscnt 0x0" ::: "memory");

    // out[n][c] = sum_s (v[idx_s][c] + peb[s][c]) * wgt[s][c/16]  (3 ch per lane)
    #pragma unroll
    for (int j = 0; j < 3; ++j) {
        int c = lane + 32 * j;
        int g = c >> 4;
        float acc3 = 0.f;
        for (int s = 0; s < NBR; ++s) {
            int nbs = nbl[wave][s];
            int rowv = nbs < 0 ? 0 : nbs;
            float val = bf2f(vbuf[(size_t)rowv * CCH + c]) + peb[s * CCH + c];
            acc3 += val * wl[s * GRP + g];
        }
        out[(size_t)n * CCH + c] = acc3;
    }
}

// ---------------------------------------------------------------------------
extern "C" void kernel_launch(void* const* d_in, const int* in_sizes, int n_in,
                              void* d_out, int out_size, void* d_ws, size_t ws_size,
                              hipStream_t stream) {
    const float* feat   = (const float*)d_in[0];
    const float* coord  = (const float*)d_in[1];
    const float* q_w    = (const float*)d_in[2];
    const float* q_b    = (const float*)d_in[3];
    const float* q_g    = (const float*)d_in[4];
    const float* q_beta = (const float*)d_in[5];
    const float* k_w    = (const float*)d_in[6];
    const float* k_b    = (const float*)d_in[7];
    const float* k_g    = (const float*)d_in[8];
    const float* k_beta = (const float*)d_in[9];
    const float* v_w    = (const float*)d_in[10];
    const float* v_b    = (const float*)d_in[11];
    const float* p1_w   = (const float*)d_in[12];
    const float* p1_b   = (const float*)d_in[13];
    const float* p_g    = (const float*)d_in[14];
    const float* p_beta = (const float*)d_in[15];
    const float* p2_w   = (const float*)d_in[16];
    const float* p2_b   = (const float*)d_in[17];
    const float* w1_w   = (const float*)d_in[18];
    const float* w1_b   = (const float*)d_in[19];
    const float* w_g    = (const float*)d_in[20];
    const float* w_beta = (const float*)d_in[21];
    const float* w2_w   = (const float*)d_in[22];
    const float* w2_b   = (const float*)d_in[23];
    const int*   refidx = (const int*)d_in[24];

    // workspace carve-out (256B aligned slices)
    char*  ws  = (char*)d_ws;
    size_t off = 0;
    auto carve = [&](size_t bytes) -> void* {
        void* p = ws + off;
        off += (bytes + 255) & ~(size_t)255;
        return p;
    };
    bf16_t* qkvp = (bf16_t*)carve(3 * 9216 * sizeof(bf16_t));
    bf16_t* p2p  = (bf16_t*)carve(9216 * sizeof(bf16_t));
    bf16_t* w1p  = (bf16_t*)carve(3 * 512 * sizeof(bf16_t));
    bf16_t* qbuf = (bf16_t*)carve((size_t)NPTS * CCH * sizeof(bf16_t));
    bf16_t* kbuf = (bf16_t*)carve((size_t)NPTS * CCH * sizeof(bf16_t));
    bf16_t* vbuf = (bf16_t*)carve((size_t)NPTS * CCH * sizeof(bf16_t));
    (void)ws_size; (void)in_sizes; (void)n_in; (void)out_size;

    pack_weights<<<40, 256, 0, stream>>>(q_w, k_w, v_w, p2_w, w1_w, qkvp, p2p, w1p);

    int tiles = (NPTS + 15) / 16;                 // 6250
    int qkv_blocks = (tiles + 3) / 4;             // 4 waves/block
    qkv_kernel<<<qkv_blocks, 128, 0, stream>>>(feat,
        q_b, q_g, q_beta, k_b, k_g, k_beta, v_b,
        qkvp, qbuf, kbuf, vbuf);

    int attn_blocks = (NPTS + 3) / 4;             // 1 point/wave, 4 waves/block
    attn_kernel<<<attn_blocks, 128, 0, stream>>>(coord, refidx,
        p1_w, p1_b, p_g, p_beta, p2_b,
        w1_b, w_g, w_beta, w2_w, w2_b,
        p2p, w1p, qbuf, kbuf, vbuf,
        (float*)d_out);
}
